// TransformerBlock_54159537603132
// MI455X (gfx1250) — compile-verified, matrix operand order checked by reading, not run on hardware
//
#include <hip/hip_runtime.h>

typedef unsigned short u16;
typedef __attribute__((ext_vector_type(16))) __bf16 v16bf;
typedef __attribute__((ext_vector_type(8)))  float  v8f;

#define DIM     4096
#define HEADS   32
#define KVHEADS 8
#define HD      128
#define KVDIM   1024      // KVHEADS*HD
#define HIDDEN  11008
#define BSZ     4
#define SEQ     1024
#define TOK     (BSZ*SEQ) // 4096 tokens

// LDS tile geometry for the staged GEMM (elements, bf16)
#define LDS_LD    40          // 32 data + 8 pad -> 80B rows, conflict-free b128 reads
#define A_ROWS    128
#define B_ROWS    256
#define A_ELEMS   (A_ROWS * LDS_LD)          // 5120
#define B_ELEMS   (B_ROWS * LDS_LD)          // 10240
#define BUF_ELEMS (A_ELEMS + B_ELEMS)        // 15360 elems = 30720 B per buffer

// ---------------- helpers ----------------

__device__ __forceinline__ u16 f2bf(float f) {
  unsigned u = __float_as_uint(f);
  unsigned r = u + 0x7FFFu + ((u >> 16) & 1u);   // round-to-nearest-even
  return (u16)(r >> 16);
}

__device__ __forceinline__ v8f zero8() {
  v8f z = {0.f,0.f,0.f,0.f,0.f,0.f,0.f,0.f};
  return z;
}

// 16x32 bf16 WMMA fragment from global (A layout; B mirrors with lane=N).
__device__ __forceinline__ v16bf frag_ld(const u16* base, int ld, int row, int k, int lane) {
  const u16* p = base + (long)(row + (lane & 15)) * ld + k + ((lane >> 4) << 3);
  union { uint4 u[2]; v16bf v; } r;
  r.u[0] = *reinterpret_cast<const uint4*>(p);
  r.u[1] = *reinterpret_cast<const uint4*>(p + 16);
  return r.v;
}

// 16x32 bf16 fragment from an LDS tile (row-major, ld = LDS_LD elements)
__device__ __forceinline__ v16bf frag_lds(const u16* tile, int row, int lane) {
  const u16* p = tile + (row + (lane & 15)) * LDS_LD + ((lane >> 4) << 3);
  union { uint4 u[2]; v16bf v; } r;
  r.u[0] = *reinterpret_cast<const uint4*>(p);
  r.u[1] = *reinterpret_cast<const uint4*>(p + 16);
  return r.v;
}

// async 16B copy global -> LDS (ASYNCcnt-tracked)
__device__ __forceinline__ void async_cp16(unsigned lds_byte_addr, const u16* gaddr) {
  asm volatile("global_load_async_to_lds_b128 %0, %1, off"
               :: "v"(lds_byte_addr), "v"(gaddr) : "memory");
}
__device__ __forceinline__ void wait_async0() {
  asm volatile("s_wait_asynccnt 0" ::: "memory");
}

__device__ __forceinline__ float redmax16(float x) {
  x = fmaxf(x, __shfl_xor(x, 1, 32));
  x = fmaxf(x, __shfl_xor(x, 2, 32));
  x = fmaxf(x, __shfl_xor(x, 4, 32));
  x = fmaxf(x, __shfl_xor(x, 8, 32));
  return x;
}
__device__ __forceinline__ float redsum16(float x) {
  x += __shfl_xor(x, 1, 32);
  x += __shfl_xor(x, 2, 32);
  x += __shfl_xor(x, 4, 32);
  x += __shfl_xor(x, 8, 32);
  return x;
}

// ---------------- elementwise kernels ----------------

__global__ void cvt_bf16_kernel(const float* __restrict__ in, u16* __restrict__ out, long n4) {
  long i = (long)blockIdx.x * blockDim.x + threadIdx.x;
  long stride = (long)gridDim.x * blockDim.x;
  for (; i < n4; i += stride) {
    float4 v = reinterpret_cast<const float4*>(in)[i];
    ushort4 o;
    o.x = f2bf(v.x); o.y = f2bf(v.y); o.z = f2bf(v.z); o.w = f2bf(v.w);
    reinterpret_cast<ushort4*>(out)[i] = o;
  }
}

__global__ void add2_kernel(float* __restrict__ dst, const float* __restrict__ a,
                            const float* __restrict__ b, long n4) {
  long i = (long)blockIdx.x * blockDim.x + threadIdx.x;
  long stride = (long)gridDim.x * blockDim.x;
  for (; i < n4; i += stride) {
    float4 x = reinterpret_cast<const float4*>(a)[i];
    float4 y = reinterpret_cast<const float4*>(b)[i];
    float4 o; o.x = x.x + y.x; o.y = x.y + y.y; o.z = x.z + y.z; o.w = x.w + y.w;
    reinterpret_cast<float4*>(dst)[i] = o;
  }
}

__global__ void swiglu_kernel(const float* __restrict__ g, const float* __restrict__ u,
                              u16* __restrict__ out, long n4) {
  long i = (long)blockIdx.x * blockDim.x + threadIdx.x;
  long stride = (long)gridDim.x * blockDim.x;
  for (; i < n4; i += stride) {
    float4 gv = reinterpret_cast<const float4*>(g)[i];
    float4 uv = reinterpret_cast<const float4*>(u)[i];
    ushort4 o;
    o.x = f2bf(gv.x / (1.f + __expf(-gv.x)) * uv.x);
    o.y = f2bf(gv.y / (1.f + __expf(-gv.y)) * uv.y);
    o.z = f2bf(gv.z / (1.f + __expf(-gv.z)) * uv.z);
    o.w = f2bf(gv.w / (1.f + __expf(-gv.w)) * uv.w);
    reinterpret_cast<ushort4*>(out)[i] = o;
  }
}

// ---------------- RMSNorm (one block per token) ----------------

__global__ __launch_bounds__(256) void rmsnorm_kernel(const float* __restrict__ x,
                                                      const float* __restrict__ w,
                                                      u16* __restrict__ out) {
  int t = blockIdx.x;
  const float4* xp = reinterpret_cast<const float4*>(x + (long)t * DIM);
  const float4* wp = reinterpret_cast<const float4*>(w);
  float s = 0.f;
  for (int i = threadIdx.x; i < DIM / 4; i += 256) {
    float4 v = xp[i];
    s += v.x * v.x + v.y * v.y + v.z * v.z + v.w * v.w;
  }
  __shared__ float red[256];
  red[threadIdx.x] = s;
  __syncthreads();
  for (int o = 128; o > 0; o >>= 1) {
    if (threadIdx.x < o) red[threadIdx.x] += red[threadIdx.x + o];
    __syncthreads();
  }
  float inv = rsqrtf(red[0] / (float)DIM + 1e-5f);
  ushort4* op = reinterpret_cast<ushort4*>(out + (long)t * DIM);
  for (int i = threadIdx.x; i < DIM / 4; i += 256) {
    float4 v = xp[i]; float4 g = wp[i];
    ushort4 o;
    o.x = f2bf(v.x * inv * g.x); o.y = f2bf(v.y * inv * g.y);
    o.z = f2bf(v.z * inv * g.z); o.w = f2bf(v.w * inv * g.w);
    op[i] = o;
  }
}

// ---------------- RoPE (fused f32 -> bf16) ----------------

__global__ void rope_kernel(const float* __restrict__ in, u16* __restrict__ out,
                            const float* __restrict__ cs, const float* __restrict__ sn, int nh) {
  long total = (long)TOK * nh * (HD / 2);
  long idx = (long)blockIdx.x * blockDim.x + threadIdx.x;
  if (idx >= total) return;
  int  j    = (int)(idx & (HD / 2 - 1));
  long rest = idx >> 6;
  int  hh   = (int)(rest % nh);
  long t    = rest / nh;
  int  s    = (int)(t & (SEQ - 1));   // start_pos == 0
  long base = (t * nh + hh) * HD + 2 * j;
  float a = in[base], b = in[base + 1];
  float c = cs[s * (HD / 2) + j], si = sn[s * (HD / 2) + j];
  unsigned lo = f2bf(a * c - b * si);
  unsigned hi = f2bf(a * si + b * c);
  *reinterpret_cast<unsigned*>(out + base) = lo | (hi << 16);
}

// ---------------- V transpose: [b,s,kvh,d] f32 -> [b,kvh,d,s] bf16 ----------------

__global__ void vtrans_kernel(const float* __restrict__ v, u16* __restrict__ vt) {
  long total = (long)BSZ * KVHEADS * HD * SEQ;
  long idx = (long)blockIdx.x * blockDim.x + threadIdx.x;
  if (idx >= total) return;
  int s = (int)(idx & (SEQ - 1));
  int d = (int)((idx >> 10) & (HD - 1));
  int h = (int)((idx >> 17) & (KVHEADS - 1));
  int b = (int)(idx >> 20);
  vt[idx] = f2bf(v[((long)(b * SEQ + s)) * KVDIM + h * HD + d]);
}

// ---------------- bf16 WMMA GEMM with async-LDS double buffering ----------------
// C[M,N] f32 = A[M,K] * W[N,K]^T. Block = 256 threads = 8 waves.
// Block tile 128x256; per K-step the 128x32 A tile and 256x32 B tile are DMA'd
// Global->LDS with GLOBAL_LOAD_ASYNC_TO_LDS_B128 (double buffered, ASYNCcnt),
// then all 8 waves read conflict-free padded fragments from LDS.

__global__ __launch_bounds__(256) void gemm_bf16_kernel(const u16* __restrict__ A,
                                                        const u16* __restrict__ B,
                                                        float* __restrict__ C,
                                                        int M, int N, int K) {
  __shared__ __align__(16) u16 lds[2 * BUF_ELEMS];   // 60 KB

  int tid  = threadIdx.x;
  int lane = tid & 31;
  int wave = tid >> 5;
  int m0 = (wave & 1) * 64;          // wave tile within block (M)
  int n0 = (wave >> 1) * 64;         // wave tile within block (N)

  const u16* Abase = A + (long)blockIdx.y * 128 * K;
  const u16* Bbase = B + (long)blockIdx.x * 256 * K;
  unsigned lbase = (unsigned)(unsigned long long)(void*)lds;  // LDS byte offset

  // stage one K-step tile (A: 512 chunks, B: 1024 chunks of 16B) into buffer `buf`
  auto stage = [&](int k, int buf) {
    unsigned base = lbase + (unsigned)buf * (BUF_ELEMS * 2);
#pragma unroll
    for (int r = 0; r < 2; ++r) {                // A tile: 2 chunks per thread
      int id  = tid + r * 256;
      int row = id >> 2, ch = id & 3;
      async_cp16(base + (unsigned)row * (LDS_LD * 2) + (unsigned)ch * 16,
                 Abase + (long)row * K + k + ch * 8);
    }
#pragma unroll
    for (int r = 0; r < 4; ++r) {                // B tile: 4 chunks per thread
      int id  = tid + r * 256;
      int row = id >> 2, ch = id & 3;
      async_cp16(base + (unsigned)(A_ELEMS * 2) + (unsigned)row * (LDS_LD * 2) + (unsigned)ch * 16,
                 Bbase + (long)row * K + k + ch * 8);
    }
  };

  v8f acc[4][4];
#pragma unroll
  for (int i = 0; i < 4; ++i)
#pragma unroll
    for (int j = 0; j < 4; ++j) acc[i][j] = zero8();

  stage(0, 0);
  wait_async0();
  __syncthreads();

  for (int k = 0, it = 0; k < K; k += 32, ++it) {
    int buf = it & 1;
    if (k + 32 < K) stage(k + 32, buf ^ 1);      // DMA next tile while computing

    const u16* at = &lds[buf * BUF_ELEMS];
    const u16* bt = &lds[buf * BUF_ELEMS + A_ELEMS];
    v16bf a[4], b[4];
#pragma unroll
    for (int i = 0; i < 4; ++i) a[i] = frag_lds(at, m0 + i * 16, lane);
#pragma unroll
    for (int j = 0; j < 4; ++j) b[j] = frag_lds(bt, n0 + j * 16, lane);
#pragma unroll
    for (int i = 0; i < 4; ++i)
#pragma unroll
      for (int j = 0; j < 4; ++j)
        acc[i][j] = __builtin_amdgcn_wmma_f32_16x16x32_bf16(
            false, a[i], false, b[j], (short)0, acc[i][j], false, false);

    wait_async0();        // my next-tile DMAs landed
    __syncthreads();      // everyone's landed; safe to read next buffer / reuse this one
  }

  int ro = (lane >> 4) << 3;
  int co = lane & 15;
  long gm = (long)blockIdx.y * 128;
  long gn = (long)blockIdx.x * 256;
#pragma unroll
  for (int i = 0; i < 4; ++i)
#pragma unroll
    for (int j = 0; j < 4; ++j)
#pragma unroll
      for (int v = 0; v < 8; ++v)
        C[(gm + m0 + i * 16 + ro + v) * N + (gn + n0 + j * 16 + co)] = acc[i][j][v];
}

// ---------------- Flash attention (causal, GQA) ----------------
// grid.x = BSZ*HEADS, grid.y = SEQ/128; 8 waves/block, each wave owns 16 query rows.

__global__ __launch_bounds__(256) void attn_kernel(const u16* __restrict__ Q,
                                                   const u16* __restrict__ Km,
                                                   const u16* __restrict__ VT,
                                                   u16* __restrict__ O) {
  int lane = threadIdx.x & 31;
  int wave = threadIdx.x >> 5;
  int bh = blockIdx.x;
  int b = bh >> 5, h = bh & 31, kvh = h >> 2;
  int q0 = blockIdx.y * 128 + wave * 16;

  const u16* qb = Q  + (long)b * SEQ * DIM   + (long)h   * HD;
  const u16* kb = Km + (long)b * SEQ * KVDIM + (long)kvh * HD;
  const u16* vb = VT + ((long)(b * KVHEADS + kvh) * HD) * SEQ;

  v16bf qf[4];
#pragma unroll
  for (int d = 0; d < 4; ++d) qf[d] = frag_ld(qb, DIM, q0, d * 32, lane);

  v8f o[8];
  float m[8], l[8];
#pragma unroll
  for (int d = 0; d < 8; ++d) o[d] = zero8();
#pragma unroll
  for (int v = 0; v < 8; ++v) { m[v] = -1e30f; l[v] = 0.f; }

  __shared__ __align__(16) u16 lp[8][16 * 32];   // per-wave P staging (16x32 bf16)

  int ro = (lane >> 4) << 3;
  int co = lane & 15;
  const float scale = 0.08838834764831845f;      // 1/sqrt(128)
  int kvend = ((q0 + 16 + 31) >> 5) << 5;        // causal: cover kv <= q0+15

  for (int kv = 0; kv < kvend; kv += 32) {
    v8f s0 = zero8(), s1 = zero8();
#pragma unroll
    for (int d = 0; d < 4; ++d) {
      v16bf b0 = frag_ld(kb, KVDIM, kv,      d * 32, lane);
      v16bf b1 = frag_ld(kb, KVDIM, kv + 16, d * 32, lane);
      s0 = __builtin_amdgcn_wmma_f32_16x16x32_bf16(false, qf[d], false, b0, (short)0, s0, false, false);
      s1 = __builtin_amdgcn_wmma_f32_16x16x32_bf16(false, qf[d], false, b1, (short)0, s1, false, false);
    }

    float al[8];
#pragma unroll
    for (int v = 0; v < 8; ++v) {
      int qrow = q0 + ro + v;
      float x0 = s0[v] * scale + ((kv + co)      > qrow ? -1e9f : 0.f);
      float x1 = s1[v] * scale + ((kv + 16 + co) > qrow ? -1e9f : 0.f);
      float mx = redmax16(fmaxf(x0, x1));
      float mn = fmaxf(m[v], mx);
      al[v] = __expf(m[v] - mn);
      float p0 = __expf(x0 - mn);
      float p1 = __expf(x1 - mn);
      l[v] = l[v] * al[v] + redsum16(p0 + p1);
      m[v] = mn;
      lp[wave][(ro + v) * 32 + co]      = f2bf(p0);
      lp[wave][(ro + v) * 32 + 16 + co] = f2bf(p1);
    }
    asm volatile("s_wait_dscnt 0" ::: "memory");

    union { uint4 u[2]; v16bf v; } pf;
    const u16* pp = &lp[wave][(lane & 15) * 32 + ((lane >> 4) << 3)];
    pf.u[0] = *reinterpret_cast<const uint4*>(pp);
    pf.u[1] = *reinterpret_cast<const uint4*>(pp + 16);

#pragma unroll
    for (int dt = 0; dt < 8; ++dt) {
      v16bf vf = frag_ld(vb, SEQ, dt * 16, kv, lane);
#pragma unroll
      for (int v = 0; v < 8; ++v) o[dt][v] *= al[v];
      o[dt] = __builtin_amdgcn_wmma_f32_16x16x32_bf16(false, pf.v, false, vf, (short)0, o[dt], false, false);
    }
  }

  float inv[8];
#pragma unroll
  for (int v = 0; v < 8; ++v) inv[v] = 1.f / l[v];
  u16* ob = O + (long)b * SEQ * DIM + (long)h * HD;
#pragma unroll
  for (int dt = 0; dt < 8; ++dt)
#pragma unroll
    for (int v = 0; v < 8; ++v)
      ob[(long)(q0 + ro + v) * DIM + dt * 16 + co] = f2bf(o[dt][v] * inv[v]);
}

// ---------------- launch ----------------

extern "C" void kernel_launch(void* const* d_in, const int* in_sizes, int n_in,
                              void* d_out, int out_size, void* d_ws, size_t ws_size,
                              hipStream_t stream) {
  (void)in_sizes; (void)n_in; (void)out_size; (void)ws_size;
  const float* x      = (const float*)d_in[0];
  const float* wq     = (const float*)d_in[1];
  const float* wk     = (const float*)d_in[2];
  const float* wv     = (const float*)d_in[3];
  const float* wo     = (const float*)d_in[4];
  const float* w1     = (const float*)d_in[5];
  const float* w2     = (const float*)d_in[6];
  const float* w3     = (const float*)d_in[7];
  const float* anw    = (const float*)d_in[8];
  const float* fnw    = (const float*)d_in[9];
  const float* fcos   = (const float*)d_in[11];
  const float* fsin   = (const float*)d_in[12];
  float* out = (float*)d_out;

  char* ws = (char*)d_ws;
  size_t off = 0;
  auto carve = [&](size_t bytes) -> void* {
    off = (off + 255) & ~(size_t)255;
    void* p = ws + off;
    off += bytes;
    return p;
  };

  const long NWQ = (long)DIM * DIM, NWK = (long)KVDIM * DIM, NW1 = (long)HIDDEN * DIM;
  u16* wq_b  = (u16*)carve(NWQ * 2);
  u16* wk_b  = (u16*)carve(NWK * 2);
  u16* wv_b  = (u16*)carve(NWK * 2);
  u16* wo_b  = (u16*)carve(NWQ * 2);
  u16* w1_b  = (u16*)carve(NW1 * 2);
  u16* w2_b  = (u16*)carve(NW1 * 2);
  u16* w3_b  = (u16*)carve(NW1 * 2);
  u16* h_b   = (u16*)carve((long)TOK * DIM * 2);
  float* q_f = (float*)carve((long)TOK * DIM * 4);
  float* k_f = (float*)carve((long)TOK * KVDIM * 4);
  float* v_f = (float*)carve((long)TOK * KVDIM * 4);
  u16* q_b   = (u16*)carve((long)TOK * DIM * 2);
  u16* k_b   = (u16*)carve((long)TOK * KVDIM * 2);
  u16* vt_b  = (u16*)carve((long)TOK * KVDIM * 2);
  u16* at_b  = (u16*)carve((long)TOK * DIM * 2);
  float* tmp = (float*)carve((long)TOK * DIM * 4);     // reused: attn@wo, then ffn out
  u16* g_b   = (u16*)carve((long)TOK * DIM * 2);
  float* gate = (float*)carve((long)TOK * HIDDEN * 4);
  float* up   = (float*)carve((long)TOK * HIDDEN * 4);
  u16* hu_b  = (u16*)carve((long)TOK * HIDDEN * 2);

  auto cvt = [&](const float* src, u16* dst, long n) {
    long n4 = n / 4;
    int blocks = (int)((n4 + 255) / 256);
    if (blocks > 16384) blocks = 16384;
    cvt_bf16_kernel<<<blocks, 256, 0, stream>>>(src, dst, n4);
  };

  // 1) weights -> bf16
  cvt(wq, wq_b, NWQ); cvt(wk, wk_b, NWK); cvt(wv, wv_b, NWK); cvt(wo, wo_b, NWQ);
  cvt(w1, w1_b, NW1); cvt(w2, w2_b, NW1); cvt(w3, w3_b, NW1);

  // 2) h = rmsnorm(x) -> bf16
  rmsnorm_kernel<<<TOK, 256, 0, stream>>>(x, anw, h_b);

  // 3) QKV projections (async-staged WMMA GEMM)
  gemm_bf16_kernel<<<dim3(DIM / 256,   TOK / 128), 256, 0, stream>>>(h_b, wq_b, q_f, TOK, DIM,   DIM);
  gemm_bf16_kernel<<<dim3(KVDIM / 256, TOK / 128), 256, 0, stream>>>(h_b, wk_b, k_f, TOK, KVDIM, DIM);
  gemm_bf16_kernel<<<dim3(KVDIM / 256, TOK / 128), 256, 0, stream>>>(h_b, wv_b, v_f, TOK, KVDIM, DIM);

  // 4) RoPE (-> bf16) and V transpose (-> bf16 [b,kvh,d,s])
  {
    long nq = (long)TOK * HEADS * (HD / 2);
    rope_kernel<<<(int)((nq + 255) / 256), 256, 0, stream>>>(q_f, q_b, fcos, fsin, HEADS);
    long nk = (long)TOK * KVHEADS * (HD / 2);
    rope_kernel<<<(int)((nk + 255) / 256), 256, 0, stream>>>(k_f, k_b, fcos, fsin, KVHEADS);
    long nv = (long)TOK * KVDIM;
    vtrans_kernel<<<(int)((nv + 255) / 256), 256, 0, stream>>>(v_f, vt_b);
  }

  // 5) flash attention -> attn bf16 [b,s,h,d]
  attn_kernel<<<dim3(BSZ * HEADS, SEQ / 128), 256, 0, stream>>>(q_b, k_b, vt_b, at_b);

  // 6) h_out = x + attn @ wo^T   (h_out lives in d_out)
  gemm_bf16_kernel<<<dim3(DIM / 256, TOK / 128), 256, 0, stream>>>(at_b, wo_b, tmp, TOK, DIM, DIM);
  {
    long n4 = (long)TOK * DIM / 4;
    add2_kernel<<<16384, 256, 0, stream>>>(out, x, tmp, n4);
  }

  // 7) FFN: g = rmsnorm(h_out); gate/up GEMMs; swiglu; down GEMM; residual
  rmsnorm_kernel<<<TOK, 256, 0, stream>>>(out, fnw, g_b);
  gemm_bf16_kernel<<<dim3(HIDDEN / 256, TOK / 128), 256, 0, stream>>>(g_b, w1_b, gate, TOK, HIDDEN, DIM);
  gemm_bf16_kernel<<<dim3(HIDDEN / 256, TOK / 128), 256, 0, stream>>>(g_b, w3_b, up,   TOK, HIDDEN, DIM);
  {
    long n4 = (long)TOK * HIDDEN / 4;
    swiglu_kernel<<<16384, 256, 0, stream>>>(gate, up, hu_b, n4);
  }
  gemm_bf16_kernel<<<dim3(DIM / 256, TOK / 128), 256, 0, stream>>>(hu_b, w2_b, tmp, TOK, DIM, HIDDEN);
  {
    long n4 = (long)TOK * DIM / 4;
    add2_kernel<<<16384, 256, 0, stream>>>(out, out, tmp, n4);
  }
}
